// SwinTransformerBlock_54795192762631
// MI455X (gfx1250) — compile-verified, matrix operand order checked by reading, not run on hardware
//
#include <hip/hip_runtime.h>
#include <hip/hip_bf16.h>

typedef __attribute__((ext_vector_type(16))) __bf16 v16bf;
typedef __attribute__((ext_vector_type(8)))  __bf16 v8bf;
typedef __attribute__((ext_vector_type(8)))  float  v8f;
typedef __attribute__((ext_vector_type(4)))  unsigned int u32x4;
typedef __attribute__((ext_vector_type(8)))  int i32x8;
typedef __attribute__((ext_vector_type(4)))  int i32x4;
typedef unsigned short u16;
typedef unsigned int   u32;
typedef unsigned long long u64;

#define NTOK  (8 * 12288)   // B*D*N tokens
#define CDIM  256
#define NPIX  12288
#define HIDN  1024

#if defined(__has_builtin)
#if __has_builtin(__builtin_amdgcn_tensor_load_to_lds) && __has_builtin(__builtin_amdgcn_s_wait_tensorcnt)
#define HAVE_TDM 1
#endif
#endif

// ---------- helpers ----------

__device__ __forceinline__ u16 f2bf(float f) {
    u32 u = __float_as_uint(f);
    u32 r = u + 0x7FFFu + ((u >> 16) & 1u);   // round-to-nearest-even
    return (u16)(r >> 16);
}

// Load a 16x32 bf16 A-fragment (or B^T fragment) from row-major [rows][ld] u16 data.
// Lane L: row = row0 + (L&15); K-halves (L>>4)*8 .. +7 and 16+(L>>4)*8 .. +7,
// matching the CDNA5 16-bit A-matrix VGPR striping.
__device__ __forceinline__ v16bf frag_ld(const u16* base, int ld, int row0, int k0) {
    int lane = threadIdx.x & 31;
    int r    = row0 + (lane & 15);
    int kh   = (lane >> 4) << 3;
    const u16* p = base + (long)r * ld + k0 + kh;
    v8bf h0 = *(const v8bf*)(p);
    v8bf h1 = *(const v8bf*)(p + 16);
    return __builtin_shufflevector(h0, h1, 0,1,2,3,4,5,6,7,8,9,10,11,12,13,14,15);
}

__device__ __forceinline__ v8f wmma_bf(v16bf a, v16bf b, v8f c) {
    return __builtin_amdgcn_wmma_f32_16x16x32_bf16(false, a, false, b, (short)0, c,
                                                   false, false);
}

// Software-pipelined 16-row x 128-col GEMM slice: KITERS K-chunks of 32.
// Stage kc+1's A fragment and 8 B fragments are loaded BEFORE stage kc's wmmas,
// forcing double-buffered registers so loads overlap WMMA execution
// (staggered s_wait_loadcnt instead of wait-0 per wmma).
template<int KITERS>
__device__ __forceinline__ void gemm_pipe(const u16* Ab, int lda, int arow, int kA,
                                          const u16* BT, int ldb, int col0, int kB,
                                          v8f (&acc)[8]) {
    v16bf a0 = frag_ld(Ab, lda, arow, kA);
    v16bf b0[8];
#pragma unroll
    for (int ct = 0; ct < 8; ++ct) b0[ct] = frag_ld(BT, ldb, col0 + ct * 16, kB);
#pragma unroll
    for (int kc = 0; kc < KITERS; ++kc) {
        v16bf a1 = {};
        v16bf b1[8];
        if (kc + 1 < KITERS) {
            int ka = kA + (kc + 1) * 32;
            int kb = kB + (kc + 1) * 32;
            a1 = frag_ld(Ab, lda, arow, ka);
#pragma unroll
            for (int ct = 0; ct < 8; ++ct) b1[ct] = frag_ld(BT, ldb, col0 + ct * 16, kb);
        }
#pragma unroll
        for (int ct = 0; ct < 8; ++ct) acc[ct] = wmma_bf(a0, b0[ct], acc[ct]);
        a0 = a1;
#pragma unroll
        for (int ct = 0; ct < 8; ++ct) b0[ct] = b1[ct];
    }
}

// ---------- weight repack: fp32 (in,out) -> bf16 [out][in] (B^T) ----------

__global__ void k_wt_transpose(const float* __restrict__ w, u16* __restrict__ wt,
                               int IN, int OUT) {
    int idx = blockIdx.x * blockDim.x + threadIdx.x;
    if (idx >= IN * OUT) return;
    int o = idx / IN, i = idx % IN;
    wt[idx] = f2bf(w[(long)i * OUT + o]);
}

// conv_w (O,I,K=9) -> bf16 [o][k*256 + i]  (B^T with flattened K-dim = k*C + ci)
__global__ void k_wt_conv(const float* __restrict__ w, u16* __restrict__ wt) {
    int idx = blockIdx.x * blockDim.x + threadIdx.x;
    if (idx >= 256 * 2304) return;
    int o = idx / 2304, rkk = idx % 2304;
    int k = rkk >> 8, i = rkk & 255;
    wt[idx] = f2bf(w[((long)o * 256 + i) * 9 + k]);
}

// ---------- layernorm: one wave per token, wave32 shuffle reduction ----------

__global__ void __launch_bounds__(256)
k_layernorm(const float* __restrict__ src, const float* __restrict__ g,
            const float* __restrict__ bt, u16* __restrict__ dst) {
    int wave = threadIdx.x >> 5;
    int lane = threadIdx.x & 31;
    int tok  = blockIdx.x * 8 + wave;
    const float* row = src + (long)tok * CDIM + lane * 8;
    float4 a = *(const float4*)(row);
    float4 b = *(const float4*)(row + 4);
    float y[8] = {a.x, a.y, a.z, a.w, b.x, b.y, b.z, b.w};
    float s = 0.f, q = 0.f;
#pragma unroll
    for (int j = 0; j < 8; ++j) { s += y[j]; q += y[j] * y[j]; }
#pragma unroll
    for (int m = 1; m < 32; m <<= 1) {
        s += __shfl_xor(s, m, 32);
        q += __shfl_xor(q, m, 32);
    }
    float mu  = s * (1.0f / 256.0f);
    float var = q * (1.0f / 256.0f) - mu * mu;
    float rs  = rsqrtf(var + 1e-5f);
    int c0 = lane * 8;
    u32 pack[4];
#pragma unroll
    for (int j = 0; j < 4; ++j) {
        float lo = (y[2 * j]     - mu) * rs * g[c0 + 2 * j]     + bt[c0 + 2 * j];
        float hi = (y[2 * j + 1] - mu) * rs * g[c0 + 2 * j + 1] + bt[c0 + 2 * j + 1];
        pack[j] = (u32)f2bf(lo) | ((u32)f2bf(hi) << 16);
    }
    *(uint4*)(dst + (long)tok * CDIM + c0) = *(uint4*)pack;
}

// ---------- conv1d (K=9, same pad) as implicit GEMM, TDM-staged halo tile ----------

__global__ void __launch_bounds__(256)
k_conv(const u16* __restrict__ A, const u16* __restrict__ BT,
       const float* __restrict__ bias, u16* __restrict__ out) {
    __shared__ u16 lds[72 * 256];   // 64 tokens + 8 halo rows, 36.8 KB
    int t0  = blockIdx.x * 64;
    int img = t0 / NPIX;
    int lo  = img * NPIX, hi = lo + NPIX;
    int tid = threadIdx.x;

#ifdef HAVE_TDM
    // One wave issues a single 2D TDM transfer: 72 rows x 512B, contiguous
    // (data_size = 8B, tile 64x72, line stride 64). Halo under/overflow stays
    // inside 4KB buffer guards; invalid rows are zeroed below.
    if (tid < 32) {
        u64 ga  = (u64)(uintptr_t)(A + (long)(t0 - 4) * CDIM);
        u32 la  = (u32)(uintptr_t)(&lds[0]);   // low 32 bits of LDS aperture = LDS offset
        u32x4 g0;
        g0.x = 1u;                                   // count=1, user descriptor
        g0.y = la;                                   // lds_addr
        g0.z = (u32)ga;                              // global_addr[31:0]
        g0.w = ((u32)(ga >> 32) & 0x01FFFFFFu) | 0x80000000u;  // addr[56:32] | type=2
        i32x8 g1 = { 0x00030000,                     // data_size=3 (8B), no flags
                     0x00400000,                     // tensor_dim0 = 64 (in 8B units)
                     0x00480000,                     // tensor_dim1 = 72
                     0x00400000,                     // tile_dim0 = 64
                     72,                             // tile_dim1 = 72, tile_dim2 = 0
                     64,                             // tensor_dim0_stride = 64
                     0, 0 };
        i32x4 gz = { 0, 0, 0, 0 };
#if __clang_major__ >= 23
        i32x8 gz8 = { 0, 0, 0, 0, 0, 0, 0, 0 };
        __builtin_amdgcn_tensor_load_to_lds(g0, g1, gz, gz, gz8, 0);
#else
        __builtin_amdgcn_tensor_load_to_lds(g0, g1, gz, gz, 0);
#endif
        __builtin_amdgcn_s_wait_tensorcnt(0);
    }
    __syncthreads();
    {   // zero halo rows that fall outside this image (8 rows x 512B = 256 x 16B)
        int row8 = tid >> 5;
        int row  = (row8 < 4) ? row8 : 64 + row8;
        int t    = t0 + row - 4;
        if (t < lo || t >= hi) {
            int col8 = (tid & 31) << 3;
            *(uint4*)(&lds[row * 256 + col8]) = make_uint4(0u, 0u, 0u, 0u);
        }
    }
    __syncthreads();
#else
#pragma unroll
    for (int p = 0; p < 9; ++p) {                 // 2304 x 16B chunks
        int chunk = p * 256 + tid;
        int row   = chunk >> 5;
        int col8  = (chunk & 31) << 3;
        int t     = t0 + row - 4;
        uint4 val = make_uint4(0u, 0u, 0u, 0u);
        if (t >= lo && t < hi) val = *(const uint4*)(A + (long)t * CDIM + col8);
        *(uint4*)(&lds[row * 256 + col8]) = val;
    }
    __syncthreads();
#endif

    int w = tid >> 5, lane = tid & 31;
    int rg = w & 3, ch = w >> 2;
    v8f zero = {};
    v8f acc[8];
#pragma unroll
    for (int i = 0; i < 8; ++i) acc[i] = zero;

    for (int k = 0; k < 9; ++k) {
        // A from LDS (tap k selected by row offset), B from L2-resident weights.
        gemm_pipe<8>(lds, 256, 16 * rg + k, 0, BT, 2304, ch * 128, k * 256, acc);
    }
    int rowadd = (lane >> 4) << 3;
#pragma unroll
    for (int ct = 0; ct < 8; ++ct) {
        int col = ch * 128 + ct * 16 + (lane & 15);
        float bc = bias[col];
#pragma unroll
        for (int r = 0; r < 8; ++r) {
            int t = t0 + 16 * rg + r + rowadd;
            out[(long)t * CDIM + col] = f2bf(acc[ct][r] + bc);
        }
    }
}

// ---------- qkv GEMM (Ntok x 256) @ (256 x 768), head-friendly output ----------

__global__ void __launch_bounds__(256)
k_qkv(const u16* __restrict__ A, const u16* __restrict__ BT,
      const float* __restrict__ bias, u16* __restrict__ qkv) {
    int t0 = blockIdx.x * 64;
    int tid = threadIdx.x, w = tid >> 5, lane = tid & 31;
    int rg = w & 3, ch = w >> 2;
    int rowadd = (lane >> 4) << 3;
    v8f zero = {};
    for (int cc = 0; cc < 3; ++cc) {
        int colbase = ch * 384 + cc * 128;
        v8f acc[8];
#pragma unroll
        for (int i = 0; i < 8; ++i) acc[i] = zero;
        gemm_pipe<8>(A, 256, t0 + 16 * rg, 0, BT, 256, colbase, 0, acc);
#pragma unroll
        for (int ct = 0; ct < 8; ++ct) {
            int c = colbase + ct * 16 + (lane & 15);
            float bb = bias[c];
            int sect = c >> 8, hh = (c & 255) >> 5, dd = c & 31;
#pragma unroll
            for (int r = 0; r < 8; ++r) {
                int t = t0 + 16 * rg + r + rowadd;
                int win = t >> 6, m = t & 63;
                long base = (long)win * 49152 + hh * 6144;
                long addr = (sect == 0) ? base + m * 32 + dd
                          : (sect == 1) ? base + 2048 + m * 32 + dd
                                        : base + 4096 + dd * 64 + m;   // v stored transposed
                qkv[addr] = f2bf(acc[ct][r] + bb);
            }
        }
    }
}

// ---------- fused window attention: block=window, wave=head ----------

__global__ void __launch_bounds__(256)
k_attn(const u16* __restrict__ qkv, u16* __restrict__ out) {
    __shared__ u16 lds_p[8 * 64 * 64];            // 64 KB: per-head P staging
    int win = blockIdx.x;
    int tid = threadIdx.x, h = tid >> 5, lane = tid & 31;
    const u16* qp = qkv + (long)win * 49152 + h * 6144;
    const u16* kp = qp + 2048;
    const u16* vt = qp + 4096;                    // v^T (32 x 64)
    v8f zero = {};

    v16bf aq[4], bk[4];
#pragma unroll
    for (int i = 0; i < 4; ++i) {
        aq[i] = frag_ld(qp, 32, i * 16, 0);       // K = hd = 32: one chunk
        bk[i] = frag_ld(kp, 32, i * 16, 0);       // k rows serve as B^T
    }
    v8f p[4][4];
#pragma unroll
    for (int mi = 0; mi < 4; ++mi)
#pragma unroll
        for (int ni = 0; ni < 4; ++ni)
            p[mi][ni] = wmma_bf(aq[mi], bk[ni], zero);

    const float scale = 0.17677669529663689f;     // 1/sqrt(32)
    float rsum[4][8];
    u16* myp = &lds_p[h * 4096];
    int rowadd = (lane >> 4) << 3;
#pragma unroll
    for (int mi = 0; mi < 4; ++mi) {
#pragma unroll
        for (int r = 0; r < 8; ++r) {
            float mx = p[mi][0][r];
#pragma unroll
            for (int ni = 1; ni < 4; ++ni) mx = fmaxf(mx, p[mi][ni][r]);
            mx = fmaxf(mx, __shfl_xor(mx, 1, 32));
            mx = fmaxf(mx, __shfl_xor(mx, 2, 32));
            mx = fmaxf(mx, __shfl_xor(mx, 4, 32));
            mx = fmaxf(mx, __shfl_xor(mx, 8, 32));
            float s = 0.f;
            int row = mi * 16 + r + rowadd;
#pragma unroll
            for (int ni = 0; ni < 4; ++ni) {
                float e = __expf((p[mi][ni][r] - mx) * scale);
                s += e;
                myp[row * 64 + ni * 16 + (lane & 15)] = f2bf(e);
            }
            s += __shfl_xor(s, 1, 32);
            s += __shfl_xor(s, 2, 32);
            s += __shfl_xor(s, 4, 32);
            s += __shfl_xor(s, 8, 32);
            rsum[mi][r] = s;
        }
    }
    // P @ V  (LDS ops from the same wave are in-order: RAW through LDS is safe)
#pragma unroll
    for (int mi = 0; mi < 4; ++mi) {
#pragma unroll
        for (int nj = 0; nj < 2; ++nj) {
            v8f o = zero;
#pragma unroll
            for (int kc = 0; kc < 2; ++kc) {
                v16bf ap = frag_ld(myp, 64, mi * 16, kc * 32);
                v16bf bv = frag_ld(vt,  64, nj * 16, kc * 32);
                o = wmma_bf(ap, bv, o);
            }
#pragma unroll
            for (int r = 0; r < 8; ++r) {
                int t = win * 64 + mi * 16 + r + rowadd;
                int c = h * 32 + nj * 16 + (lane & 15);
                out[(long)t * CDIM + c] = f2bf(o[r] / rsum[mi][r]);
            }
        }
    }
}

// ---------- proj GEMM + residual -> x1 (fp32) ----------

__global__ void __launch_bounds__(256)
k_proj(const u16* __restrict__ A, const u16* __restrict__ BT,
       const float* __restrict__ bias, const float* __restrict__ x,
       float* __restrict__ x1) {
    int t0 = blockIdx.x * 64;
    int tid = threadIdx.x, w = tid >> 5, lane = tid & 31;
    int rg = w & 3, ch = w >> 2;
    int rowadd = (lane >> 4) << 3;
    v8f zero = {};
    v8f acc[8];
#pragma unroll
    for (int i = 0; i < 8; ++i) acc[i] = zero;
    gemm_pipe<8>(A, 256, t0 + 16 * rg, 0, BT, 256, ch * 128, 0, acc);
#pragma unroll
    for (int ct = 0; ct < 8; ++ct) {
        int c = ch * 128 + ct * 16 + (lane & 15);
        float bb = bias[c];
#pragma unroll
        for (int r = 0; r < 8; ++r) {
            int t = t0 + 16 * rg + r + rowadd;
            long idx = (long)t * CDIM + c;
            x1[idx] = x[idx] + acc[ct][r] + bb;
        }
    }
}

// ---------- fc1 GEMM + exact GELU ----------

__global__ void __launch_bounds__(256)
k_fc1(const u16* __restrict__ A, const u16* __restrict__ BT,
      const float* __restrict__ bias, u16* __restrict__ hid) {
    int t0 = blockIdx.x * 64;
    int tid = threadIdx.x, w = tid >> 5, lane = tid & 31;
    int rg = w & 3, ch = w >> 2;
    int rowadd = (lane >> 4) << 3;
    v8f zero = {};
    for (int cc = 0; cc < 4; ++cc) {
        int colbase = ch * 512 + cc * 128;
        v8f acc[8];
#pragma unroll
        for (int i = 0; i < 8; ++i) acc[i] = zero;
        gemm_pipe<8>(A, 256, t0 + 16 * rg, 0, BT, 256, colbase, 0, acc);
#pragma unroll
        for (int ct = 0; ct < 8; ++ct) {
            int c = colbase + ct * 16 + (lane & 15);
            float bb = bias[c];
#pragma unroll
            for (int r = 0; r < 8; ++r) {
                int t = t0 + 16 * rg + r + rowadd;
                float v = acc[ct][r] + bb;
                float ge = 0.5f * v * (1.0f + erff(v * 0.70710678118654752f));
                hid[(long)t * HIDN + c] = f2bf(ge);
            }
        }
    }
}

// ---------- fc2 GEMM + residual -> output (fp32) ----------

__global__ void __launch_bounds__(256)
k_fc2(const u16* __restrict__ A, const u16* __restrict__ BT,
      const float* __restrict__ bias, const float* __restrict__ x1,
      float* __restrict__ out) {
    int t0 = blockIdx.x * 64;
    int tid = threadIdx.x, w = tid >> 5, lane = tid & 31;
    int rg = w & 3, ch = w >> 2;
    int rowadd = (lane >> 4) << 3;
    v8f zero = {};
    v8f acc[8];
#pragma unroll
    for (int i = 0; i < 8; ++i) acc[i] = zero;
    for (int kb = 0; kb < 4; ++kb) {   // K=1024 in four pipelined 256-chunks
        gemm_pipe<8>(A, HIDN, t0 + 16 * rg, kb * 256, BT, HIDN, ch * 128, kb * 256, acc);
    }
#pragma unroll
    for (int ct = 0; ct < 8; ++ct) {
        int c = ch * 128 + ct * 16 + (lane & 15);
        float bb = bias[c];
#pragma unroll
        for (int r = 0; r < 8; ++r) {
            int t = t0 + 16 * rg + r + rowadd;
            long idx = (long)t * CDIM + c;
            out[idx] = x1[idx] + acc[ct][r] + bb;
        }
    }
}

// ---------- launch ----------

extern "C" void kernel_launch(void* const* d_in, const int* in_sizes, int n_in,
                              void* d_out, int out_size, void* d_ws, size_t ws_size,
                              hipStream_t stream) {
    const float* x      = (const float*)d_in[0];
    const float* ln1_g  = (const float*)d_in[1];
    const float* ln1_b  = (const float*)d_in[2];
    const float* conv_w = (const float*)d_in[3];
    const float* conv_b = (const float*)d_in[4];
    const float* qkv_w  = (const float*)d_in[5];
    const float* qkv_b  = (const float*)d_in[6];
    const float* proj_w = (const float*)d_in[7];
    const float* proj_b = (const float*)d_in[8];
    const float* ln2_g  = (const float*)d_in[9];
    const float* ln2_b  = (const float*)d_in[10];
    const float* fc1_w  = (const float*)d_in[11];
    const float* fc1_b  = (const float*)d_in[12];
    const float* fc2_w  = (const float*)d_in[13];
    const float* fc2_b  = (const float*)d_in[14];

    char* ws = (char*)d_ws;
    const size_t GUARD   = 4096;                           // TDM halo under/overflow guard
    const size_t SZ_HLN  = (size_t)NTOK * CDIM * 2;        // 50.3 MB
    const size_t SZ_QKV  = (size_t)NTOK * 768 * 2;         // 151 MB
    const size_t SZ_X1   = (size_t)NTOK * CDIM * 4;        // 100 MB
    const size_t SZ_HID  = (size_t)NTOK * HIDN * 2;        // 201 MB
    u16*   hln   = (u16*)(ws + GUARD);
    u16*   hconv = (u16*)(ws + GUARD + SZ_HLN + GUARD);
    u16*   qkv   = (u16*)(ws + 2 * GUARD + 2 * SZ_HLN);
    float* x1    = (float*)(ws + 2 * GUARD + 2 * SZ_HLN + SZ_QKV);
    u16*   hid   = (u16*)(ws + 2 * GUARD + 2 * SZ_HLN + SZ_QKV + SZ_X1);
    char*  wb    = ws + 2 * GUARD + 2 * SZ_HLN + SZ_QKV + SZ_X1 + SZ_HID;
    u16* wconv = (u16*)(wb);
    u16* wqkv  = (u16*)(wb + 1179648);
    u16* wproj = (u16*)(wb + 1179648 + 393216);
    u16* wfc1  = (u16*)(wb + 1179648 + 393216 + 131072);
    u16* wfc2  = (u16*)(wb + 1179648 + 393216 + 131072 + 524288);
    u16* attn_out = hln;     // reuse: hln dead after k_conv
    u16* ln2out   = hconv;   // reuse: hconv dead after k_qkv

    k_wt_conv<<<2304, 256, 0, stream>>>(conv_w, wconv);
    k_wt_transpose<<<(256 * 768 + 255) / 256, 256, 0, stream>>>(qkv_w, wqkv, 256, 768);
    k_wt_transpose<<<(256 * 256 + 255) / 256, 256, 0, stream>>>(proj_w, wproj, 256, 256);
    k_wt_transpose<<<(256 * 1024 + 255) / 256, 256, 0, stream>>>(fc1_w, wfc1, 256, 1024);
    k_wt_transpose<<<(1024 * 256 + 255) / 256, 256, 0, stream>>>(fc2_w, wfc2, 1024, 256);

    k_layernorm<<<NTOK / 8, 256, 0, stream>>>(x, ln1_g, ln1_b, hln);
    k_conv<<<NTOK / 64, 256, 0, stream>>>(hln, wconv, conv_b, hconv);
    k_qkv<<<NTOK / 64, 256, 0, stream>>>(hconv, wqkv, qkv_b, qkv);
    k_attn<<<NTOK / 64, 256, 0, stream>>>(qkv, attn_out);
    k_proj<<<NTOK / 64, 256, 0, stream>>>(attn_out, wproj, proj_b, x, x1);
    k_layernorm<<<NTOK / 8, 256, 0, stream>>>(x1, ln2_g, ln2_b, ln2out);
    k_fc1<<<NTOK / 64, 256, 0, stream>>>(ln2out, wfc1, fc1_b, hid);
    k_fc2<<<NTOK / 64, 256, 0, stream>>>(hid, wfc2, fc2_b, x1, (float*)d_out);
}
